// TransMIL_48567490183290
// MI455X (gfx1250) — compile-verified, matrix operand order checked by reading, not run on hardware
//
#include <hip/hip_runtime.h>

// ---------------- model constants ----------------
#define B_     8
#define N_     4096
#define CH     512          // model dim
#define INCH   1024
#define NP1    4097         // tokens incl. cls
#define PAD_   255          // front zero-pad for nystrom
#define NPAD   4352         // NP1 + PAD_
#define LM     17           // tokens per landmark (NPAD / 256)
#define MM     256          // landmarks
#define HEADS  8
#define DH     64
#define SCALE_ 0.125f       // DH^-0.5

// ---------------- WMMA types (probe-verified signatures) ----------------
typedef __attribute__((ext_vector_type(16))) __bf16 v16bf;
typedef __attribute__((ext_vector_type(8)))  float  v8f;
typedef __attribute__((ext_vector_type(4)))  float  v4f;

__device__ __forceinline__ __bf16 f2bf(float x) { return (__bf16)x; }

// fragment element position for K index k (0..31) inside the 32-element
// per-column LDS strip: order [k0..7][k16..23][k8..15][k24..31]
__device__ __forceinline__ int kperm(int k) {
    const int g = k >> 3;                       // 0..3
    return (((g & 1) << 4) | ((g & 2) << 2)) + (k & 7);   // base {0,16,8,24}
}

// ======================================================================
// Generic batched WMMA GEMM:  C = epi( scale * A[M,K] @ B[K,N] )
//   epiMode 0: C = scale*acc (+bias) (+Res) (relu?)
//   epiMode 1: C = diagAlpha*I - acc
// A fp32 row-major (lda); B fp32 row-major [K,N] (ldb) or, if btrans,
// stored [N,K] row-major (compute A @ B^T).
// Block: 256 threads = 8 waves. Block tile 256(M) x 64(N); each wave
// owns 32 rows (2 A frags) x 64 cols (4 B frags) = 8 WMMAs / K-step.
// B tile staged in LDS as bf16, pre-permuted to fragment order.
// K must be a multiple of 32 (uses: 1024, 512, 256, 64).
// OOB rows/cols handled by branchless index clamping: garbage only
// reaches D rows/cols that the guarded epilogue never stores.
// ======================================================================
__global__ __launch_bounds__(256)
void wmma_bgemm_kernel(const float* __restrict__ A, long sAb, int lda,
                       const float* __restrict__ Bm, long sBb, int ldb, int btrans,
                       float* __restrict__ C, long sCb, int ldc,
                       const float* __restrict__ Res, long sRb,
                       const float* __restrict__ bias,
                       int M, int N, int K,
                       float scale, float diagAlpha, int epiMode, int doRelu)
{
    const int tid  = threadIdx.x;
    const int wave = tid >> 5;
    const int lane = tid & 31;
    const int half = lane >> 4;       // 0 or 1 (lane group)
    const int l16  = lane & 15;
    const int rowBase = blockIdx.x * 256 + wave * 32;
    const int colBase = blockIdx.y * 64;

    A  += (long)blockIdx.z * sAb;
    Bm += (long)blockIdx.z * sBb;
    C  += (long)blockIdx.z * sCb;
    const float* Rp = Res ? (Res + (long)blockIdx.z * sRb) : (const float*)0;

    // clamped A rows (garbage rows never stored)
    int aRow0 = rowBase + l16;        if (aRow0 > M - 1) aRow0 = M - 1;
    int aRow1 = rowBase + 16 + l16;   if (aRow1 > M - 1) aRow1 = M - 1;
    const float* Arow0 = A + (long)aRow0 * lda;
    const float* Arow1 = A + (long)aRow1 * lda;

    __shared__ __bf16 Blds[64 * 32];  // [colLocal][32 permuted K] = 4 KB

    const bool colsFull = (colBase + 64) <= N;   // block-uniform

    v8f acc[2][4] = {};

    for (int k0 = 0; k0 < K; k0 += 32) {
        __syncthreads();              // protect LDS from previous readers
        // ---- stage B tile (32 K x 64 cols) into LDS as permuted bf16 ----
        if (btrans) {
            // thread handles 8 consecutive K for one column
            const int idx  = tid * 8;
            const int nloc = idx >> 5;            // 0..63
            const int k    = idx & 31;            // 0,8,16,24
            int ng = colBase + nloc; if (ng > N - 1) ng = N - 1;
            const float* src = Bm + (long)ng * ldb + k0 + k;
            const v4f b0 = *(const v4f*)src;
            const v4f b1 = *(const v4f*)(src + 4);
            const int pb = kperm(k);              // k multiple of 8 -> run stays in group
            __bf16* dst = &Blds[nloc * 32 + pb];
            #pragma unroll
            for (int i = 0; i < 4; ++i) { dst[i] = f2bf(b0[i]); dst[4 + i] = f2bf(b1[i]); }
        } else {
            // thread handles 8 consecutive columns for one K row
            const int idx  = tid * 8;
            const int k    = idx >> 6;            // 0..31
            const int nloc = idx & 63;            // multiple of 8
            const int pk   = kperm(k);
            if (colsFull) {
                const float* src = Bm + (long)(k0 + k) * ldb + colBase + nloc;
                const v4f b0 = *(const v4f*)src;
                const v4f b1 = *(const v4f*)(src + 4);
                #pragma unroll
                for (int i = 0; i < 4; ++i) {
                    Blds[(nloc + i) * 32 + pk]     = f2bf(b0[i]);
                    Blds[(nloc + 4 + i) * 32 + pk] = f2bf(b1[i]);
                }
            } else {
                #pragma unroll
                for (int i = 0; i < 8; ++i) {
                    int ngc = colBase + nloc + i; if (ngc > N - 1) ngc = N - 1;
                    Blds[(nloc + i) * 32 + pk] = f2bf(Bm[(long)(k0 + k) * ldb + ngc]);
                }
            }
        }
        __syncthreads();

        // ---- A fragments: 2 contiguous 8-float runs each, vectorized ----
        const int r0 = k0 + half * 8;
        const int r1 = k0 + 16 + half * 8;
        __builtin_prefetch(Arow0 + k0 + 32, 0, 1);
        __builtin_prefetch(Arow1 + k0 + 32, 0, 1);

        v16bf af0, af1;
        {
            const v4f a0 = *(const v4f*)(Arow0 + r0);
            const v4f a1 = *(const v4f*)(Arow0 + r0 + 4);
            const v4f a2 = *(const v4f*)(Arow0 + r1);
            const v4f a3 = *(const v4f*)(Arow0 + r1 + 4);
            #pragma unroll
            for (int i = 0; i < 4; ++i) {
                af0[i] = f2bf(a0[i]); af0[4 + i] = f2bf(a1[i]);
                af0[8 + i] = f2bf(a2[i]); af0[12 + i] = f2bf(a3[i]);
            }
        }
        {
            const v4f a0 = *(const v4f*)(Arow1 + r0);
            const v4f a1 = *(const v4f*)(Arow1 + r0 + 4);
            const v4f a2 = *(const v4f*)(Arow1 + r1);
            const v4f a3 = *(const v4f*)(Arow1 + r1 + 4);
            #pragma unroll
            for (int i = 0; i < 4; ++i) {
                af1[i] = f2bf(a0[i]); af1[4 + i] = f2bf(a1[i]);
                af1[8 + i] = f2bf(a2[i]); af1[12 + i] = f2bf(a3[i]);
            }
        }

        #pragma unroll
        for (int t = 0; t < 4; ++t) {
            const int nloc = t * 16 + l16;
            const v16bf bf = *(const v16bf*)&Blds[nloc * 32 + half * 16];
            acc[0][t] = __builtin_amdgcn_wmma_f32_16x16x32_bf16(
                            false, af0, false, bf, (short)0, acc[0][t], false, false);
            acc[1][t] = __builtin_amdgcn_wmma_f32_16x16x32_bf16(
                            false, af1, false, bf, (short)0, acc[1][t], false, false);
        }
    }

    // C/D layout: vgpr r, lanes0-15 -> M=r,N=l16 ; lanes16-31 -> M=r+8
    #pragma unroll
    for (int u = 0; u < 2; ++u) {
        #pragma unroll
        for (int t = 0; t < 4; ++t) {
            const int gc = colBase + t * 16 + l16;
            if (gc >= N) continue;
            #pragma unroll
            for (int r = 0; r < 8; ++r) {
                const int gr = rowBase + u * 16 + r + half * 8;
                if (gr >= M) continue;
                const float v = acc[u][t][r];
                float o;
                if (epiMode == 1) {
                    o = ((gr == gc) ? diagAlpha : 0.f) - v;
                } else {
                    o = v * scale;
                    if (bias) o += bias[gc];
                    if (Rp)   o += Rp[(long)gr * ldc + gc];
                    if (doRelu) o = fmaxf(o, 0.f);
                }
                C[(long)gr * ldc + gc] = o;
            }
        }
    }
}

// ======================================================================
// PPEG: depthwise 7x7 + 5x5 + 3x3 + identity on 64x64 grid (channels-last)
// writes into h rows [1 .. 4096]
// ======================================================================
__global__ __launch_bounds__(256)
void ppeg_kernel(const float* __restrict__ h0,
                 const float* __restrict__ k7, const float* __restrict__ b7,
                 const float* __restrict__ k5, const float* __restrict__ b5,
                 const float* __restrict__ k3, const float* __restrict__ b3,
                 float* __restrict__ h)
{
    long t = (long)blockIdx.x * 256 + threadIdx.x;   // 8*4096*512
    const int c = (int)(t % CH); long r = t / CH;
    const int n = (int)(r % N_); const int b = (int)(r / N_);
    const int y = n >> 6, x = n & 63;
    const float* base = h0 + ((long)b * N_) * CH + c;
    float acc = base[(long)n * CH] + b7[c] + b5[c] + b3[c];
    const float* w7 = k7 + c * 49;
    for (int dy = 0; dy < 7; ++dy) { int yy = y + dy - 3; if ((unsigned)yy >= 64u) continue;
        for (int dx = 0; dx < 7; ++dx) { int xx = x + dx - 3; if ((unsigned)xx >= 64u) continue;
            acc += base[(long)((yy << 6) + xx) * CH] * w7[dy * 7 + dx]; } }
    const float* w5 = k5 + c * 25;
    for (int dy = 0; dy < 5; ++dy) { int yy = y + dy - 2; if ((unsigned)yy >= 64u) continue;
        for (int dx = 0; dx < 5; ++dx) { int xx = x + dx - 2; if ((unsigned)xx >= 64u) continue;
            acc += base[(long)((yy << 6) + xx) * CH] * w5[dy * 5 + dx]; } }
    const float* w3 = k3 + c * 9;
    for (int dy = 0; dy < 3; ++dy) { int yy = y + dy - 1; if ((unsigned)yy >= 64u) continue;
        for (int dx = 0; dx < 3; ++dx) { int xx = x + dx - 1; if ((unsigned)xx >= 64u) continue;
            acc += base[(long)((yy << 6) + xx) * CH] * w3[dy * 3 + dx]; } }
    h[(((long)b * NP1) + 1 + n) * CH + c] = acc;
}

__global__ void cls_kernel(const float* __restrict__ cls, float* __restrict__ h)
{
    int t = blockIdx.x * 256 + threadIdx.x;  // 8*512
    int c = t & 511; int b = t >> 9;
    h[(long)b * NP1 * CH + c] = cls[c];
}

// ======================================================================
// LayerNorm over 512 with front zero-padding
// ======================================================================
__global__ __launch_bounds__(256)
void ln_pad_kernel(const float* __restrict__ in, long sInB,
                   const float* __restrict__ g, const float* __restrict__ bb,
                   float* __restrict__ out, long sOutB, int rowsOut, int pad)
{
    const int orow  = blockIdx.x % rowsOut;
    const int batch = blockIdx.x / rowsOut;
    const int tid = threadIdx.x;
    float* op = out + (long)batch * sOutB + (long)orow * CH;
    if (orow < pad) { op[tid] = 0.f; op[tid + 256] = 0.f; return; }
    const float* ip = in + (long)batch * sInB + (long)(orow - pad) * CH;
    const float x0 = ip[tid], x1 = ip[tid + 256];
    __shared__ float red[256];
    red[tid] = x0 + x1; __syncthreads();
    for (int st = 128; st > 0; st >>= 1) { if (tid < st) red[tid] += red[tid + st]; __syncthreads(); }
    const float mu = red[0] * (1.f / 512.f);
    __syncthreads();
    const float d0 = x0 - mu, d1 = x1 - mu;
    red[tid] = d0 * d0 + d1 * d1; __syncthreads();
    for (int st = 128; st > 0; st >>= 1) { if (tid < st) red[tid] += red[tid + st]; __syncthreads(); }
    const float inv = rsqrtf(red[0] * (1.f / 512.f) + 1e-5f);
    op[tid]       = d0 * inv * g[tid]       + bb[tid];
    op[tid + 256] = d1 * inv * g[tid + 256] + bb[tid + 256];
}

// ======================================================================
// Landmark means: q_l (scaled) and k_l from qkv buffer [B,NPAD,1536]
// ======================================================================
__global__ __launch_bounds__(256)
void landmark_kernel(const float* __restrict__ qkv,
                     float* __restrict__ ql, float* __restrict__ kl)
{
    long t = (long)blockIdx.x * 256 + threadIdx.x;   // 64*256*64
    const int d = (int)(t & 63); long r = t >> 6;
    const int mm = (int)(r & 255); r >>= 8;
    const int hh = (int)(r & 7); const int b = (int)(r >> 3);
    const float* base = qkv + (long)b * NPAD * 1536 + hh * DH + d;
    float sq = 0.f, sk = 0.f;
    const int i0 = mm * LM;
    for (int j = 0; j < LM; ++j) {
        const float* p = base + (long)(i0 + j) * 1536;
        sq += p[0]; sk += p[CH];
    }
    const long o = (((long)b * HEADS + hh) * MM + mm) * DH + d;
    ql[o] = sq * (SCALE_ / (float)LM);
    kl[o] = sk * (1.f / (float)LM);
}

// row softmax, row length 256, one block per row
__global__ __launch_bounds__(256)
void softmax256_kernel(float* __restrict__ X)
{
    float* p = X + (long)blockIdx.x * 256;
    const int tid = threadIdx.x;
    const float v = p[tid];
    __shared__ float red[256];
    red[tid] = v; __syncthreads();
    for (int st = 128; st > 0; st >>= 1) { if (tid < st) red[tid] = fmaxf(red[tid], red[tid + st]); __syncthreads(); }
    const float m = red[0]; __syncthreads();
    const float e = __expf(v - m);
    red[tid] = e; __syncthreads();
    for (int st = 128; st > 0; st >>= 1) { if (tid < st) red[tid] += red[tid + st]; __syncthreads(); }
    p[tid] = e / red[0];
}

// ---- pinv helpers ----
__global__ __launch_bounds__(256)
void pinv_sums_kernel(const float* __restrict__ a2, float* __restrict__ RS, float* __restrict__ CS)
{
    int t = blockIdx.x * 256 + threadIdx.x;   // 64*256
    const int i = t & 255; const int bh = t >> 8;
    const float* Xp = a2 + (long)bh * 65536;
    float rs = 0.f, cs = 0.f;
    for (int j = 0; j < 256; ++j) { rs += fabsf(Xp[i * 256 + j]); cs += fabsf(Xp[j * 256 + i]); }
    RS[t] = rs; CS[t] = cs;
}

__global__ __launch_bounds__(256)
void pinv_scale_kernel(const float* __restrict__ RS, const float* __restrict__ CS, float* __restrict__ scl)
{
    const int tid = threadIdx.x;
    float m1 = 0.f, m2 = 0.f;
    for (int i = tid; i < 64 * 256; i += 256) { m1 = fmaxf(m1, RS[i]); m2 = fmaxf(m2, CS[i]); }
    __shared__ float r1[256], r2[256];
    r1[tid] = m1; r2[tid] = m2; __syncthreads();
    for (int st = 128; st > 0; st >>= 1) {
        if (tid < st) { r1[tid] = fmaxf(r1[tid], r1[tid + st]); r2[tid] = fmaxf(r2[tid], r2[tid + st]); }
        __syncthreads();
    }
    if (tid == 0) scl[0] = 1.f / (r1[0] * r2[0]);
}

__global__ void zinit_kernel(const float* __restrict__ a2, const float* __restrict__ scl, float* __restrict__ z)
{
    long t = (long)blockIdx.x * 256 + threadIdx.x;  // 64*256*256
    const int j = (int)(t & 255); long r = t >> 8;
    const int i = (int)(r & 255); const int bh = (int)(r >> 8);
    z[t] = a2[((long)bh * 256 + j) * 256 + i] * scl[0];   // x^T * s
}

__global__ void diag_sub_kernel(const float* __restrict__ XZ, float* __restrict__ U, float alpha)
{
    long t = (long)blockIdx.x * 256 + threadIdx.x;
    const int j = (int)(t & 255); const int i = (int)((t >> 8) & 255);
    U[t] = ((i == j) ? alpha : 0.f) - XZ[t];
}

// ======================================================================
// a3v: flash-style  softmax(q_l @ k^T) @ v  -> kv [64,256,64]
// ======================================================================
__global__ __launch_bounds__(256)
void a3v_kernel(const float* __restrict__ qkv, const float* __restrict__ ql,
                float* __restrict__ kv)
{
    const int bh = blockIdx.x; const int b = bh >> 3; const int hh = bh & 7;
    const int wave = threadIdx.x >> 5; const int lane = threadIdx.x & 31;
    __shared__ float Kt[32 * 65];
    __shared__ float Vt[32 * 65];
    __shared__ float qrow[8 * 64];
    __shared__ float pb[8 * 33];
    const int row = blockIdx.y * 8 + wave;   // 0..255
    for (int e = threadIdx.x; e < 512; e += 256) {
        int w = e >> 6, d = e & 63;
        qrow[e] = ql[((long)bh * MM + blockIdx.y * 8 + w) * DH + d];
    }
    __syncthreads();
    const float* kb = qkv + (long)b * NPAD * 1536 + CH + hh * DH;
    const float* vb = kb + CH;
    float m = -1e30f, s = 0.f, a0 = 0.f, a1 = 0.f;
    for (int jb = 0; jb < NPAD; jb += 32) {
        for (int e = threadIdx.x; e < 2048; e += 256) {
            int j = e >> 6, d = e & 63;
            Kt[j * 65 + d] = kb[(long)(jb + j) * 1536 + d];
            Vt[j * 65 + d] = vb[(long)(jb + j) * 1536 + d];
        }
        __syncthreads();
        float lg = 0.f;
        const float* qr = qrow + wave * 64;
        for (int d = 0; d < 64; ++d) lg += qr[d] * Kt[lane * 65 + d];
        float tm = lg;
        for (int off = 16; off; off >>= 1) tm = fmaxf(tm, __shfl_xor(tm, off, 32));
        const float nm = fmaxf(m, tm);
        const float corr = __expf(m - nm);
        const float p = __expf(lg - nm);
        float ps = p;
        for (int off = 16; off; off >>= 1) ps += __shfl_xor(ps, off, 32);
        s = s * corr + ps;
        a0 *= corr; a1 *= corr;
        m = nm;
        pb[wave * 33 + lane] = p;
        __syncthreads();
        for (int j = 0; j < 32; ++j) {
            const float pj = pb[wave * 33 + j];
            a0 += pj * Vt[j * 65 + lane];
            a1 += pj * Vt[j * 65 + lane + 32];
        }
        __syncthreads();
    }
    float* op = kv + ((long)bh * MM + row) * DH;
    op[lane]      = a0 / s;
    op[lane + 32] = a1 / s;
}

// ======================================================================
// out[b,i,h*64+d] = softmax(q_i . k_l^T) @ P    (P = pinv @ a3v, [256,64])
// ======================================================================
__global__ __launch_bounds__(256)
void a1_out_kernel(const float* __restrict__ qkv, const float* __restrict__ kl,
                   const float* __restrict__ P, float* __restrict__ outc)
{
    const int bh = blockIdx.x; const int b = bh >> 3; const int hh = bh & 7;
    const int wave = threadIdx.x >> 5; const int lane = threadIdx.x & 31;
    __shared__ float tile[64 * 65];
    __shared__ float qrow[8 * 64];
    __shared__ float pb[8 * 257];
    const float* qbase = qkv + (long)b * NPAD * 1536 + hh * DH;
    const float* klb = kl + (long)bh * MM * DH;
    const float* Pb  = P  + (long)bh * MM * DH;

    for (int it = 0; it < 8; ++it) {
        const int row = blockIdx.y * 64 + it * 8 + wave;  // < 4352 always
        qrow[wave * 64 + lane]      = qbase[(long)row * 1536 + lane] * SCALE_;
        qrow[wave * 64 + lane + 32] = qbase[(long)row * 1536 + lane + 32] * SCALE_;
        __syncthreads();
        float lg[8];
        for (int c = 0; c < 4; ++c) {
            for (int e = threadIdx.x; e < 4096; e += 256) {
                int j = e >> 6, d = e & 63;
                tile[j * 65 + d] = klb[(long)(c * 64 + j) * DH + d];
            }
            __syncthreads();
            #pragma unroll
            for (int u = 0; u < 2; ++u) {
                const int j = u * 32 + lane;
                float sdot = 0.f;
                for (int d = 0; d < 64; ++d) sdot += qrow[wave * 64 + d] * tile[j * 65 + d];
                lg[c * 2 + u] = sdot;
            }
            __syncthreads();
        }
        float mx = lg[0];
        #pragma unroll
        for (int jj = 1; jj < 8; ++jj) mx = fmaxf(mx, lg[jj]);
        for (int off = 16; off; off >>= 1) mx = fmaxf(mx, __shfl_xor(mx, off, 32));
        float ssum = 0.f, pr[8];
        #pragma unroll
        for (int jj = 0; jj < 8; ++jj) { pr[jj] = __expf(lg[jj] - mx); ssum += pr[jj]; }
        for (int off = 16; off; off >>= 1) ssum += __shfl_xor(ssum, off, 32);
        #pragma unroll
        for (int jj = 0; jj < 8; ++jj)
            pb[wave * 257 + (jj >> 1) * 64 + (jj & 1) * 32 + lane] = pr[jj];
        __syncthreads();
        float o0 = 0.f, o1 = 0.f;
        for (int c = 0; c < 4; ++c) {
            for (int e = threadIdx.x; e < 4096; e += 256) {
                int j = e >> 6, d = e & 63;
                tile[j * 65 + d] = Pb[(long)(c * 64 + j) * DH + d];
            }
            __syncthreads();
            for (int j = 0; j < 64; ++j) {
                const float pj = pb[wave * 257 + c * 64 + j];
                o0 += pj * tile[j * 65 + lane];
                o1 += pj * tile[j * 65 + lane + 32];
            }
            __syncthreads();
        }
        const float inv = 1.f / ssum;
        float* op = outc + ((long)b * NPAD + row) * CH + hh * DH;
        op[lane]      = o0 * inv;
        op[lane + 32] = o1 * inv;
        __syncthreads();
    }
}

// residual depthwise conv over sequence (33 taps), v read from qkv buffer
__global__ __launch_bounds__(256)
void resadd_kernel(const float* __restrict__ qkv, const float* __restrict__ resk,
                   float* __restrict__ outc)
{
    long t = (long)blockIdx.x * 256 + threadIdx.x;   // 8*4352*512
    const int cd = (int)(t % CH); long r = t / CH;
    const int i = (int)(r % NPAD); const int b = (int)(r / NPAD);
    const int hh = cd >> 6;
    const float* vcol = qkv + (long)b * NPAD * 1536 + 1024 + cd;
    float acc = 0.f;
    for (int tt = 0; tt < 33; ++tt) {
        const int ii = i + tt - 16;
        if ((unsigned)ii < (unsigned)NPAD) acc += vcol[(long)ii * 1536] * resk[hh * 33 + tt];
    }
    outc[t] += acc;
}

// ======================================================================
// host side
// ======================================================================
static void bgemm(hipStream_t s, const float* A, long sAb, int lda,
                  const float* B, long sBb, int ldb, int btrans,
                  float* C, long sCb, int ldc,
                  const float* Res, long sRb, const float* bias,
                  int M, int N, int K, float scale, float diagAlpha,
                  int epi, int relu, int batch)
{
    dim3 g((M + 255) / 256, (N + 63) / 64, batch);
    wmma_bgemm_kernel<<<g, 256, 0, s>>>(A, sAb, lda, B, sBb, ldb, btrans,
                                        C, sCb, ldc, Res, sRb, bias,
                                        M, N, K, scale, diagAlpha, epi, relu);
}

extern "C" void kernel_launch(void* const* d_in, const int* in_sizes, int n_in,
                              void* d_out, int out_size, void* d_ws, size_t ws_size,
                              hipStream_t stream)
{
    (void)in_sizes; (void)n_in; (void)out_size; (void)ws_size;
    const float* x     = (const float*)d_in[0];
    const float* W1    = (const float*)d_in[1];
    const float* b1    = (const float*)d_in[2];
    const float* cls   = (const float*)d_in[3];
    const float* k7    = (const float*)d_in[4];  const float* b7 = (const float*)d_in[5];
    const float* k5    = (const float*)d_in[6];  const float* b5 = (const float*)d_in[7];
    const float* k3    = (const float*)d_in[8];  const float* b3 = (const float*)d_in[9];
    const float* ln1g  = (const float*)d_in[10]; const float* ln1b = (const float*)d_in[11];
    const float* qkv1  = (const float*)d_in[12];
    const float* out1w = (const float*)d_in[13]; const float* out1b = (const float*)d_in[14];
    const float* res1k = (const float*)d_in[15];
    const float* ln2g  = (const float*)d_in[16]; const float* ln2b = (const float*)d_in[17];
    const float* qkv2  = (const float*)d_in[18];
    const float* out2w = (const float*)d_in[19]; const float* out2b = (const float*)d_in[20];
    const float* res2k = (const float*)d_in[21];
    const float* lnfg  = (const float*)d_in[22]; const float* lnfb = (const float*)d_in[23];
    const float* W2    = (const float*)d_in[24]; const float* b2 = (const float*)d_in[25];

    // ---- workspace bump allocator (floats, 256B aligned) -> ~540 MB ----
    size_t off = 0;
    auto alloc = [&](size_t nf) {
        float* p = (float*)((char*)d_ws + off);
        off += ((nf * sizeof(float) + 255) / 256) * 256;
        return p;
    };
    float* h      = alloc((size_t)B_ * NP1 * CH);     // residual stream
    float* buf0   = alloc((size_t)B_ * N_ * CH);      // gemm1 / ppeg input
    float* lnbuf  = alloc((size_t)B_ * NPAD * CH);    // LN(+pad); later reused as outc
    float* qkvbuf = alloc((size_t)B_ * NPAD * 1536);
    float* qlbuf  = alloc((size_t)64 * MM * DH);
    float* klbuf  = alloc((size_t)64 * MM * DH);
    float* a2buf  = alloc((size_t)64 * MM * MM);
    float* zbuf   = alloc((size_t)64 * MM * MM);
    float* z2buf  = alloc((size_t)64 * MM * MM);
    float* xzbuf  = alloc((size_t)64 * MM * MM);
    float* ubuf   = alloc((size_t)64 * MM * MM);
    float* wbuf   = alloc((size_t)64 * MM * MM);
    float* kvbuf  = alloc((size_t)64 * MM * DH);
    float* Pbuf   = alloc((size_t)64 * MM * DH);
    float* RS     = alloc((size_t)64 * MM);
    float* CS     = alloc((size_t)64 * MM);
    float* scl    = alloc(64);
    float* flnbuf = alloc((size_t)B_ * CH);
    float* outc   = lnbuf;    // alias: lnbuf dead after qkv GEMM

    // 1) h0 = relu(x @ W1 + b1)
    bgemm(stream, x, 0, INCH, W1, 0, CH, 0, buf0, 0, CH,
          nullptr, 0, b1, B_ * N_, CH, INCH, 1.f, 0.f, 0, 1, 1);
    // 2) PPEG -> h rows 1..4096 ; 3) cls -> h row 0
    ppeg_kernel<<<65536, 256, 0, stream>>>(buf0, k7, b7, k5, b5, k3, b3, h);
    cls_kernel<<<16, 256, 0, stream>>>(cls, h);

    auto layer = [&](const float* lng, const float* lnb, const float* Wqkv,
                     const float* Wout, const float* bout, const float* resk)
    {
        ln_pad_kernel<<<B_ * NPAD, 256, 0, stream>>>(h, (long)NP1 * CH, lng, lnb,
                                                     lnbuf, (long)NPAD * CH, NPAD, PAD_);
        bgemm(stream, lnbuf, 0, CH, Wqkv, 0, 1536, 0, qkvbuf, 0, 1536,
              nullptr, 0, nullptr, B_ * NPAD, 1536, CH, 1.f, 0.f, 0, 0, 1);
        landmark_kernel<<<4096, 256, 0, stream>>>(qkvbuf, qlbuf, klbuf);
        // a2 = softmax(q_l @ k_l^T)
        bgemm(stream, qlbuf, MM * DH, DH, klbuf, MM * DH, DH, 1, a2buf, MM * MM, MM,
              nullptr, 0, nullptr, MM, MM, DH, 1.f, 0.f, 0, 0, 64);
        softmax256_kernel<<<64 * MM, 256, 0, stream>>>(a2buf);
        // Moore-Penrose iterative pseudoinverse
        pinv_sums_kernel<<<64, 256, 0, stream>>>(a2buf, RS, CS);
        pinv_scale_kernel<<<1, 256, 0, stream>>>(RS, CS, scl);
        zinit_kernel<<<16384, 256, 0, stream>>>(a2buf, scl, zbuf);
        float* zc = zbuf; float* zn = z2buf;
        for (int it = 0; it < 6; ++it) {
            bgemm(stream, a2buf, 65536, MM, zc, 65536, MM, 0, xzbuf, 65536, MM,
                  nullptr, 0, nullptr, MM, MM, MM, 1.f, 0.f, 0, 0, 64);      // XZ = x@z
            diag_sub_kernel<<<16384, 256, 0, stream>>>(xzbuf, ubuf, 7.f);    // U = 7I - XZ
            bgemm(stream, xzbuf, 65536, MM, ubuf, 65536, MM, 0, wbuf, 65536, MM,
                  nullptr, 0, nullptr, MM, MM, MM, 1.f, 15.f, 1, 0, 64);     // W = 15I - XZ@U
            bgemm(stream, xzbuf, 65536, MM, wbuf, 65536, MM, 0, ubuf, 65536, MM,
                  nullptr, 0, nullptr, MM, MM, MM, 1.f, 13.f, 1, 0, 64);     // Y = 13I - XZ@W
            bgemm(stream, zc, 65536, MM, ubuf, 65536, MM, 0, zn, 65536, MM,
                  nullptr, 0, nullptr, MM, MM, MM, 0.25f, 0.f, 0, 0, 64);    // z = 0.25 z@Y
            float* tmp = zc; zc = zn; zn = tmp;
        }
        // a3v (flash), P = pinv @ a3v, out = softmax(q k_l^T) @ P, + res conv
        a3v_kernel<<<dim3(64, 32), 256, 0, stream>>>(qkvbuf, qlbuf, kvbuf);
        bgemm(stream, zc, 65536, MM, kvbuf, MM * DH, DH, 0, Pbuf, MM * DH, DH,
              nullptr, 0, nullptr, MM, DH, MM, 1.f, 0.f, 0, 0, 64);
        a1_out_kernel<<<dim3(64, 68), 256, 0, stream>>>(qkvbuf, klbuf, Pbuf, outc);
        resadd_kernel<<<69632, 256, 0, stream>>>(qkvbuf, resk, outc);
        // h += (outc[:, PAD_: , :] @ Wout + bout)
        bgemm(stream, outc + (long)PAD_ * CH, (long)NPAD * CH, CH, Wout, 0, CH, 0,
              h, (long)NP1 * CH, CH, h, (long)NP1 * CH, bout,
              NP1, CH, CH, 1.f, 0.f, 0, 0, B_);
    };

    layer(ln1g, ln1b, qkv1, out1w, out1b, res1k);
    layer(ln2g, ln2b, qkv2, out2w, out2b, res2k);

    // final: LN of token 0, then classifier
    ln_pad_kernel<<<B_, 256, 0, stream>>>(h, (long)NP1 * CH, lnfg, lnfb,
                                          flnbuf, (long)CH, 1, 0);
    bgemm(stream, flnbuf, 0, CH, W2, 0, 1000, 0, (float*)d_out, 0, 1000,
          nullptr, 0, b2, B_, 1000, CH, 1.f, 0.f, 0, 0, 1);
}
